// SE3Transformer_29008209117808
// MI455X (gfx1250) — matrix operand order repaired and checked by reading, not compiled, not run on hardware
//
#include <hip/hip_runtime.h>

#define HID 128
#define NHARM 9
#define NLAYERS 6
#define TOTAL 65536          // B*N
#define WAVES 8
#define ROWS 16              // points per wave tile
#define LNEPS 1e-5f
#define NSTAGE 36            // 9 harmonics x 4 K-chunks, 8KB each

#if defined(__gfx1250__) && __has_builtin(__builtin_amdgcn_global_load_async_to_lds_b128)
#define USE_ASYNC 1
#else
#define USE_ASYNC 0
#endif

typedef __attribute__((ext_vector_type(16))) __bf16 v16bf;
typedef __attribute__((ext_vector_type(8)))  __bf16 v8bf;
typedef __attribute__((ext_vector_type(8)))  float  v8f;
typedef int v4i __attribute__((vector_size(16)));   // builtin's pointee type

__device__ __forceinline__ void wait_async0() {
#if USE_ASYNC
#if __has_builtin(__builtin_amdgcn_s_wait_asynccnt)
  __builtin_amdgcn_s_wait_asynccnt(0);
#else
  asm volatile("s_wait_asynccnt 0x0" ::: "memory");
#endif
#endif
}

// -------------------------------------------------------------------------
// Pack Wl [6][128][128][9] f32 -> bf16 B-fragment images, one 32KB image per
// (layer t, harmonic l).  Image element index: jc*4096 + oc*512 + lane*16 + s
// holds W[c = jc*32 + (lane>>4)*16 + s][o = oc*16 + (lane&15)][l]
// (32x16 bf16 B-matrix in this code's contiguous K order: lanes 0-15 K=0..15, lanes 16-31 K=16..31,
//  two K per VGPR half, N = lane%16).
// -------------------------------------------------------------------------
__global__ void prep_pack(const float* __restrict__ Wl, __bf16* __restrict__ packed) {
  int e = blockIdx.x * 256 + threadIdx.x;
  if (e >= NLAYERS * NHARM * 16384) return;
  int within = e & 16383;
  int tl     = e >> 14;
  int l = tl % NHARM, t = tl / NHARM;
  int s    =  within        & 15;
  int lane = (within >> 4)  & 31;
  int oc   = (within >> 9)  & 7;
  int jc   =  within >> 12;
  int c = jc * 32 + (lane >> 4) * 16 + s;
  int o = oc * 16 + (lane & 15);
  packed[e] = (__bf16)Wl[((t * HID + c) * HID + o) * NHARM + l];
}

// -------------------------------------------------------------------------
// Fully fused SE(3)-pointwise network. 1 wave = 16 points, 8 waves/WG.
// -------------------------------------------------------------------------
__global__ void __launch_bounds__(256) se3_fused(
    const float* __restrict__ x,     const float* __restrict__ pos,
    const float* __restrict__ W_in,  const float* __restrict__ b_in,
    const __bf16* __restrict__ Wpk,  const float* __restrict__ bl,
    const float* __restrict__ gamma, const float* __restrict__ beta,
    const float* __restrict__ W_out, const float* __restrict__ b_out,
    float* __restrict__ out)
{
  __shared__ __align__(32) __bf16 hbuf[WAVES][ROWS][HID];   // 32 KB activations (bf16)
  __shared__ __align__(32) __bf16 wbuf[2][4096];            // 2 x 8 KB weight stages
  __shared__ float Ybuf[WAVES][ROWS][NHARM];                // 4.5 KB
  __shared__ float xsb[WAVES][ROWS];

  const int tid  = threadIdx.x;
  const int w    = tid >> 5;
  const int lane = tid & 31;
  const int rowbase = (blockIdx.x * WAVES + w) * ROWS;
  const int hig = lane >> 4;     // which half of the K-chunk this lane covers
  const int nlo = lane & 15;     // row (A) / column (B,D) index within tile

  // issue one 8KB weight stage (layer tt, stage ss = l*4+jc) into wbuf[b]
  auto issue_stage = [&](int tt, int ss, int b) {
#if USE_ASYNC
    const __bf16* g = Wpk + (size_t)(tt * NHARM + (ss >> 2)) * 16384
                          + (ss & 3) * 4096 + tid * 16;
    __bf16* d = &wbuf[b][tid * 16];
    __builtin_amdgcn_global_load_async_to_lds_b128(
        (v4i*)(g),
        (__attribute__((address_space(3))) v4i*)(d), 0, 0);
    __builtin_amdgcn_global_load_async_to_lds_b128(
        (v4i*)(g + 8),
        (__attribute__((address_space(3))) v4i*)(d + 8), 0, 0);
#endif
    (void)tt; (void)ss; (void)b;
  };

  // ---------------- spherical harmonics + input scalar ----------------
  if (lane < 16) {
    int n = rowbase + lane;
    float px = pos[3 * n], py = pos[3 * n + 1], pz = pos[3 * n + 2];
    float r  = sqrtf(px * px + py * py + pz * pz);
    float th = acosf(pz / (r + 1e-8f));
    float ph = atan2f(py, px);
    float st = sinf(th), ct = cosf(th);
    float sp = sinf(ph), cp = cosf(ph);
    float X = st * cp, Yv = st * sp, Z = ct;
    float* Yp = Ybuf[w][lane];
    Yp[0] = 0.28209479177387814f;
    Yp[1] = 0.4886025119029199f * Yv;
    Yp[2] = 0.4886025119029199f * Z;
    Yp[3] = 0.4886025119029199f * X;
    Yp[4] = 1.0925484305920792f * X * Yv;
    Yp[5] = 1.0925484305920792f * Yv * Z;
    Yp[6] = 0.31539156525252005f * (3.f * Z * Z - 1.f);
    Yp[7] = 1.0925484305920792f * X * Z;
    Yp[8] = 0.5462742152960396f * (X * X - Yv * Yv);
    xsb[w][lane] = x[n];
  }
  __syncthreads();

  // per-lane Y scalars for the A-row this lane carries
  float yf[NHARM];
  #pragma unroll
  for (int l = 0; l < NHARM; ++l) yf[l] = Ybuf[w][nlo][l];

  // ---------------- input projection: h0 = x * (Y . W_in) + b_in ----------------
  for (int idx = lane; idx < ROWS * HID; idx += 32) {
    int r = idx >> 7, o = idx & 127;
    float s = 0.f;
    #pragma unroll
    for (int l = 0; l < NHARM; ++l) s += Ybuf[w][r][l] * W_in[o * NHARM + l];
    hbuf[w][r][o] = (__bf16)(xsb[w][r] * s + b_in[o]);
  }
  __syncthreads();

#if USE_ASYNC
  issue_stage(0, 0, 0);            // prefetch first weight stage
#endif

  v8f acc[8];

  for (int t = 0; t < NLAYERS; ++t) {
    // -------- load A base fragments (16x32 bf16 layout) from h tile --------
    v16bf abase[4];
    #pragma unroll
    for (int jc = 0; jc < 4; ++jc) {
      const __bf16* hrow = &hbuf[w][nlo][0];
      v8bf lo8 = *(const v8bf*)(hrow + jc * 32 + hig * 16);      // K = hig*16 .. +7
      v8bf hi8 = *(const v8bf*)(hrow + jc * 32 + hig * 16 + 8);  // K = hig*16+8 .. +15
      abase[jc] = __builtin_shufflevector(lo8, hi8,
                    0, 1, 2, 3, 4, 5, 6, 7, 8, 9, 10, 11, 12, 13, 14, 15);
    }
    #pragma unroll
    for (int oc = 0; oc < 8; ++oc)
      #pragma unroll
      for (int q = 0; q < 8; ++q) acc[oc][q] = 0.f;

    // -------- 36 double-buffered 8KB weight stages of 8 WMMAs each --------
    for (int s = 0; s < NSTAGE; ++s) {
      const int b = s & 1;
#if USE_ASYNC
      wait_async0();               // this wave's stage-s bytes are in LDS
      __syncthreads();             // everyone's bytes in; prev compute done WG-wide
      if (s + 1 < NSTAGE)        issue_stage(t, s + 1, (s + 1) & 1);
      else if (t + 1 < NLAYERS)  issue_stage(t + 1, 0, 0);  // overlaps LN epilogue
#else
      __syncthreads();             // all waves done reading wbuf[b] (stage s-2)
      {
        const uint4* srcv = (const uint4*)(Wpk + (size_t)(t * NHARM + (s >> 2)) * 16384
                                               + (s & 3) * 4096);
        uint4* dstv = (uint4*)wbuf[b];
        dstv[tid]       = srcv[tid];
        dstv[tid + 256] = srcv[tid + 256];
      }
      __syncthreads();
#endif
      const int l = s >> 2;
      v16bf asc;                   // A rows scaled by Y[n,l] (per-lane scalar)
      #pragma unroll
      for (int e = 0; e < 16; ++e)
        asc[e] = (__bf16)((float)abase[s & 3][e] * yf[l]);
      #pragma unroll
      for (int oc = 0; oc < 8; ++oc) {
        v16bf bfrag = *(const v16bf*)(&wbuf[b][oc * 512 + lane * 16]);
        acc[oc] = __builtin_amdgcn_wmma_f32_16x16x32_bf16(
                      false, asc, false, bfrag, (short)0, acc[oc], false, false);
      }
    }

    // -------- bias, LayerNorm (in D-fragment layout), ReLU, residual --------
    float blv[8], gl[8], bt[8];
    #pragma unroll
    for (int oc = 0; oc < 8; ++oc) {
      int col = oc * 16 + nlo;
      blv[oc] = bl[t * HID + col];
      gl[oc]  = gamma[t * HID + col];
      bt[oc]  = beta[t * HID + col];
    }
    #pragma unroll
    for (int oc = 0; oc < 8; ++oc)
      #pragma unroll
      for (int q = 0; q < 8; ++q) acc[oc][q] += blv[oc];

    float sum[8], ssq[8];
    #pragma unroll
    for (int q = 0; q < 8; ++q) { sum[q] = 0.f; ssq[q] = 0.f; }
    #pragma unroll
    for (int oc = 0; oc < 8; ++oc)
      #pragma unroll
      for (int q = 0; q < 8; ++q) { float v = acc[oc][q]; sum[q] += v; ssq[q] += v * v; }
    #pragma unroll
    for (int m = 1; m < 16; m <<= 1)
      #pragma unroll
      for (int q = 0; q < 8; ++q) {
        sum[q] += __shfl_xor(sum[q], m, 32);
        ssq[q] += __shfl_xor(ssq[q], m, 32);
      }
    float mu[8], inv[8];
    #pragma unroll
    for (int q = 0; q < 8; ++q) {
      mu[q] = sum[q] * (1.f / HID);
      float var = ssq[q] * (1.f / HID) - mu[q] * mu[q];
      inv[q] = rsqrtf(var + LNEPS);
    }

    #pragma unroll
    for (int oc = 0; oc < 8; ++oc) {
      int col = oc * 16 + nlo;
      #pragma unroll
      for (int q = 0; q < 8; ++q) {
        int row = q + 8 * hig;                      // D layout: lanes>=16 hold M=q+8
        float v = (acc[oc][q] - mu[q]) * inv[q] * gl[oc] + bt[oc];
        v = fmaxf(v, 0.f);
        v += (float)hbuf[w][row][col];              // residual (read before overwrite)
        hbuf[w][row][col] = (__bf16)v;
      }
    }
    __syncthreads();
  }

  // ---------------- output projection ----------------
  if (lane < 16) {
    int r = lane;
    float a = b_out[0];
    for (int o = 0; o < HID; ++o) {
      float s = 0.f;
      #pragma unroll
      for (int l = 0; l < NHARM; ++l) s += Ybuf[w][r][l] * W_out[o * NHARM + l];
      a += (float)hbuf[w][r][o] * s;
    }
    out[rowbase + r] = a;
  }
}

extern "C" void kernel_launch(void* const* d_in, const int* in_sizes, int n_in,
                              void* d_out, int out_size, void* d_ws, size_t ws_size,
                              hipStream_t stream) {
  const float* x     = (const float*)d_in[0];
  const float* pos   = (const float*)d_in[1];
  const float* W_in  = (const float*)d_in[2];
  const float* b_in  = (const float*)d_in[3];
  const float* Wl    = (const float*)d_in[4];
  const float* bl    = (const float*)d_in[5];
  const float* gamma = (const float*)d_in[6];
  const float* beta  = (const float*)d_in[7];
  const float* W_out = (const float*)d_in[8];
  const float* b_out = (const float*)d_in[9];
  float* out = (float*)d_out;
  __bf16* Wpk = (__bf16*)d_ws;   // 6*9*16384 bf16 = 1.73 MB packed weights

  prep_pack<<<(NLAYERS * NHARM * 16384) / 256, 256, 0, stream>>>(Wl, Wpk);
  se3_fused<<<TOTAL / (WAVES * ROWS), 256, 0, stream>>>(
      x, pos, W_in, b_in, Wpk, bl, gamma, beta, W_out, b_out, out);
}
